// RumourGAT_23811298689159
// MI455X (gfx1250) — matrix operand order repaired and checked by reading, not compiled, not run on hardware
//
#include <hip/hip_runtime.h>
#include <math.h>

// ---------------- problem constants ----------------
#define NN 50000            // nodes  (50000 % 16 == 0 -> perfect WMMA tiling)
#define EE 800000           // raw edges
#define ETOT 850000         // edges + self loops
#define F1 128              // HEADS*HID = 4*32
#define HID 32
#define HEADS 4
#define NSLOPE 0.2f

typedef __attribute__((ext_vector_type(16))) _Float16 v16h;
typedef __attribute__((ext_vector_type(8)))  _Float16 v8h;
typedef __attribute__((ext_vector_type(8)))  float    v8f;

// ---------------- workspace layout (float offsets) ----------------
// phase-1 region
#define OFF_H1    0u                           // N*128
#define OFF_AS1   6400000u                     // N*4
#define OFF_AD1   6600000u
#define OFF_M1    6800000u
#define OFF_D1    7000000u
#define OFF_EX1   7200000u                     // ETOT*4
#define OFF_OUT1 10600000u                     // N*128
#define OFF_X2H  17000000u                     // N*128 halves (N*64 float slots)
#define OFF_W2T  20200000u                     // 4096 halves, W2 transposed [32][128]
// phase-2 region: aliases dead h1/out1 space
#define OFF_H2        0u                       // N*32
#define OFF_AS2 1600000u                       // N
#define OFF_AD2 1650000u
#define OFF_M2  1700000u
#define OFF_D2  1750000u
#define OFF_EX2 1800000u                       // ETOT
#define OFF_OUT2 2650000u                      // N*32
// total ~20.2M floats = ~81 MB  (fits comfortably in 192MB L2)

__device__ __forceinline__ float lrelu(float v) { return v > 0.f ? v : NSLOPE * v; }
__device__ __forceinline__ float eluf(float v)  { return v > 0.f ? v : (__expf(v) - 1.f); }

__device__ __forceinline__ void atomAddF(float* p, float v) {
#if __has_builtin(__hip_atomic_fetch_add)
  __hip_atomic_fetch_add(p, v, __ATOMIC_RELAXED, __HIP_MEMORY_SCOPE_AGENT); // -> global_atomic_add_f32
#else
  atomicAdd(p, v);
#endif
}
__device__ __forceinline__ void atomMaxF(float* p, float v) {
#if __has_builtin(__hip_atomic_fetch_max)
  __hip_atomic_fetch_max(p, v, __ATOMIC_RELAXED, __HIP_MEMORY_SCOPE_AGENT); // -> global_atomic_max_num_f32
#else
  int* ip = (int*)p;
  int old = __float_as_int(*p);
  while (__int_as_float(old) < v) {
    int assumed = old;
    old = atomicCAS(ip, assumed, __float_as_int(v));
    if (old == assumed) break;
  }
#endif
}

// edge i -> (src, dst) including implicit self loops appended after EE
__device__ __forceinline__ void edge_sd(const int* ei, int i, int& s, int& d) {
  if (i < EE) { s = ei[i]; d = ei[EE + i]; }
  else        { s = i - EE; d = i - EE; }
}

// ---------------- layer 1: x@W1 (K=5) + per-head attention dots ----------------
__global__ void k_lin1(const float* __restrict__ x, const float* __restrict__ W1,
                       const float* __restrict__ attS, const float* __restrict__ attD,
                       float* __restrict__ h1, float* __restrict__ as1, float* __restrict__ ad1) {
  int t = blockIdx.x * blockDim.x + threadIdx.x;
  if (t >= NN * HEADS) return;
  int n = t >> 2, head = t & 3;
  float xi[5];
#pragma unroll
  for (int i = 0; i < 5; ++i) xi[i] = x[n * 5 + i];
  float sa = 0.f, sd = 0.f;
  int cbase = head * HID;
#pragma unroll 4
  for (int c = 0; c < HID; ++c) {
    float acc = 0.f;
#pragma unroll
    for (int i = 0; i < 5; ++i) acc = fmaf(xi[i], W1[i * F1 + cbase + c], acc);
    h1[n * F1 + cbase + c] = acc;
    sa = fmaf(acc, attS[cbase + c], sa);
    sd = fmaf(acc, attD[cbase + c], sd);
  }
  as1[n * HEADS + head] = sa;
  ad1[n * HEADS + head] = sd;
}

__global__ void k_init1(float* __restrict__ out1, float* __restrict__ m1, float* __restrict__ d1) {
  int t = blockIdx.x * blockDim.x + threadIdx.x;
  if (t < NN * F1) out1[t] = 0.f;
  if (t < NN * HEADS) { m1[t] = -3.4e38f; d1[t] = 0.f; }
}

__global__ void k_edge1_max(const int* __restrict__ ei, const float* __restrict__ as1,
                            const float* __restrict__ ad1, float* __restrict__ m1) {
  int i = blockIdx.x * blockDim.x + threadIdx.x;
  if (i >= ETOT) return;
  int s, d; edge_sd(ei, i, s, d);
  float4 a = *(const float4*)(as1 + s * 4);
  float4 b = *(const float4*)(ad1 + d * 4);
  atomMaxF(m1 + d * 4 + 0, lrelu(a.x + b.x));
  atomMaxF(m1 + d * 4 + 1, lrelu(a.y + b.y));
  atomMaxF(m1 + d * 4 + 2, lrelu(a.z + b.z));
  atomMaxF(m1 + d * 4 + 3, lrelu(a.w + b.w));
}

__global__ void k_edge1_sum(const int* __restrict__ ei, const float* __restrict__ as1,
                            const float* __restrict__ ad1, const float* __restrict__ m1,
                            float* __restrict__ ex1, float* __restrict__ d1) {
  int i = blockIdx.x * blockDim.x + threadIdx.x;
  if (i >= ETOT) return;
  int s, d; edge_sd(ei, i, s, d);
  float4 a = *(const float4*)(as1 + s * 4);
  float4 b = *(const float4*)(ad1 + d * 4);
  float4 m = *(const float4*)(m1 + d * 4);
  float4 ex;
  ex.x = __expf(lrelu(a.x + b.x) - m.x);
  ex.y = __expf(lrelu(a.y + b.y) - m.y);
  ex.z = __expf(lrelu(a.z + b.z) - m.z);
  ex.w = __expf(lrelu(a.w + b.w) - m.w);
  *(float4*)(ex1 + i * 4) = ex;
  atomAddF(d1 + d * 4 + 0, ex.x);
  atomAddF(d1 + d * 4 + 1, ex.y);
  atomAddF(d1 + d * 4 + 2, ex.z);
  atomAddF(d1 + d * 4 + 3, ex.w);
}

// 32 threads per edge; each thread: one head's 4-channel float4 slice
__global__ void k_edge1_agg(const int* __restrict__ ei, const float* __restrict__ ex1,
                            const float* __restrict__ d1, const float* __restrict__ h1,
                            float* __restrict__ out1) {
  long long t = (long long)blockIdx.x * blockDim.x + threadIdx.x;
  if (t >= (long long)ETOT * 32) return;
  int i = (int)(t >> 5), c4 = (int)(t & 31);
  int s, d; edge_sd(ei, i, s, d);
  int head = c4 >> 3;                 // 8 slices per head
  int cb = c4 * 4;                    // channel base in [0,128)
  float alpha = ex1[i * 4 + head] / d1[d * 4 + head];
  float4 hv = *(const float4*)(h1 + s * F1 + cb);
  float* o = out1 + d * F1 + cb;
  atomAddF(o + 0, alpha * hv.x);
  atomAddF(o + 1, alpha * hv.y);
  atomAddF(o + 2, alpha * hv.z);
  atomAddF(o + 3, alpha * hv.w);
}

// bias + ELU, convert straight to f16 for the WMMA GEMM
__global__ void k_elu1(const float* __restrict__ out1, const float* __restrict__ b1,
                       _Float16* __restrict__ x2h) {
  int t = blockIdx.x * blockDim.x + threadIdx.x;
  if (t >= NN * F1) return;
  x2h[t] = (_Float16)eluf(out1[t] + b1[t & (F1 - 1)]);
}

// pack W2 [128,32] -> transposed f16 Wt [32][128] so each lane's B fragment
// (K = ks+8g..+7 and ks+16+8g..+7 at fixed column) is two contiguous 16-byte runs
__global__ void k_cvtW2(const float* __restrict__ W2, _Float16* __restrict__ Wt) {
  int t = blockIdx.x * blockDim.x + threadIdx.x;
  if (t >= F1 * HID) return;
  int k = t >> 5, n = t & 31;
  Wt[n * F1 + k] = (_Float16)W2[t];
}

// ---------------- layer 2 GEMM: [N,128] x [128,32] via v_wmma_f32_16x16x32_f16 ----
// one wave per 16x16 output tile; 4 WMMA steps over K=128.
// A fragment (16-bit A 16x32, ISA 7.12.2): lane(M=lm, group g) holds
//   K = ks+8g..ks+8g+7  and  K = ks+16+8g..+7   -> two contiguous v8h loads.
// B fragment (16-bit B 32x16): lane(N=lm, group g) holds K = ks+8g..ks+8g+7  and  K = ks+16+8g..+7
//   -> two contiguous v8h loads from the transposed weight pack.
__global__ void k_gemm2(const _Float16* __restrict__ X, const _Float16* __restrict__ Wt,
                        float* __restrict__ H2) {
  int wave = (int)((blockIdx.x * blockDim.x + threadIdx.x) >> 5);
  const int tiles = (NN / 16) * (HID / 16);     // 6250
  if (wave >= tiles) return;                    // wave-uniform: EXEC stays all-ones
  int lane = threadIdx.x & 31;
  int tn = wave & 1, tm = wave >> 1;
  int m0 = tm * 16, n0 = tn * 16;
  int lm = lane & 15, g = lane >> 4;
  const _Float16* arow = X + (m0 + lm) * F1;
  const _Float16* brow = Wt + (n0 + lm) * F1;
  v8f c = {};
#pragma unroll
  for (int ks = 0; ks < F1; ks += 32) {
    v8h a0 = *(const v8h*)(arow + ks + g * 8);        // K = ks+8g .. +7
    v8h a1 = *(const v8h*)(arow + ks + 16 + g * 8);   // K = ks+16+8g .. +7
    v16h a = __builtin_shufflevector(a0, a1, 0, 1, 2, 3, 4, 5, 6, 7,
                                             8, 9, 10, 11, 12, 13, 14, 15);
    v8h b0 = *(const v8h*)(brow + ks + g * 8);        // K = ks+8g .. +7
    v8h b1 = *(const v8h*)(brow + ks + 16 + g * 8);   // K = ks+16+8g .. +7
    v16h b = __builtin_shufflevector(b0, b1, 0, 1, 2, 3, 4, 5, 6, 7,
                                             8, 9, 10, 11, 12, 13, 14, 15);
    c = __builtin_amdgcn_wmma_f32_16x16x32_f16(false, a, false, b, (short)0, c, false, false);
  }
#pragma unroll
  for (int r = 0; r < 8; ++r)                   // D: VGPR r -> M = r + 8g, N = lm
    H2[(m0 + r + 8 * g) * HID + n0 + lm] = c[r];
}

// attention dots for layer 2 (1 head, 32 ch)
__global__ void k_att2(const float* __restrict__ h2, const float* __restrict__ attS,
                       const float* __restrict__ attD, float* __restrict__ as2,
                       float* __restrict__ ad2) {
  int n = blockIdx.x * blockDim.x + threadIdx.x;
  if (n >= NN) return;
  float sa = 0.f, sd = 0.f;
#pragma unroll 8
  for (int c = 0; c < HID; ++c) {
    float h = h2[n * HID + c];
    sa = fmaf(h, attS[c], sa);
    sd = fmaf(h, attD[c], sd);
  }
  as2[n] = sa; ad2[n] = sd;
}

__global__ void k_init2(float* __restrict__ out2, float* __restrict__ m2, float* __restrict__ d2) {
  int t = blockIdx.x * blockDim.x + threadIdx.x;
  if (t < NN * HID) out2[t] = 0.f;
  if (t < NN) { m2[t] = -3.4e38f; d2[t] = 0.f; }
}

__global__ void k_edge2_max(const int* __restrict__ ei, const float* __restrict__ as2,
                            const float* __restrict__ ad2, float* __restrict__ m2) {
  int i = blockIdx.x * blockDim.x + threadIdx.x;
  if (i >= ETOT) return;
  int s, d; edge_sd(ei, i, s, d);
  atomMaxF(m2 + d, lrelu(as2[s] + ad2[d]));
}

__global__ void k_edge2_sum(const int* __restrict__ ei, const float* __restrict__ as2,
                            const float* __restrict__ ad2, const float* __restrict__ m2,
                            float* __restrict__ ex2, float* __restrict__ d2) {
  int i = blockIdx.x * blockDim.x + threadIdx.x;
  if (i >= ETOT) return;
  int s, d; edge_sd(ei, i, s, d);
  float ex = __expf(lrelu(as2[s] + ad2[d]) - m2[d]);
  ex2[i] = ex;
  atomAddF(d2 + d, ex);
}

// 8 threads per edge, float4 slices of 32 channels
__global__ void k_edge2_agg(const int* __restrict__ ei, const float* __restrict__ ex2,
                            const float* __restrict__ d2, const float* __restrict__ h2,
                            float* __restrict__ out2) {
  long long t = (long long)blockIdx.x * blockDim.x + threadIdx.x;
  if (t >= (long long)ETOT * 8) return;
  int i = (int)(t >> 3), j = (int)(t & 7);
  int s, d; edge_sd(ei, i, s, d);
  float alpha = ex2[i] / d2[d];
  float4 hv = *(const float4*)(h2 + s * HID + j * 4);
  float* o = out2 + d * HID + j * 4;
  atomAddF(o + 0, alpha * hv.x);
  atomAddF(o + 1, alpha * hv.y);
  atomAddF(o + 2, alpha * hv.z);
  atomAddF(o + 3, alpha * hv.w);
}

// fused head: elu(out2+b2) -> relu(@Wc1+bc1) -> @Wc2+bc2
__global__ void k_head(const float* __restrict__ out2, const float* __restrict__ b2,
                       const float* __restrict__ Wc1, const float* __restrict__ bc1,
                       const float* __restrict__ Wc2, const float* __restrict__ bc2,
                       float* __restrict__ out) {
  int n = blockIdx.x * blockDim.x + threadIdx.x;
  if (n >= NN) return;
  float x3[HID];
#pragma unroll 8
  for (int c = 0; c < HID; ++c) x3[c] = eluf(out2[n * HID + c] + b2[c]);
  float acc = bc2[0];
#pragma unroll
  for (int k = 0; k < 16; ++k) {
    float h = bc1[k];
#pragma unroll 8
    for (int c = 0; c < HID; ++c) h = fmaf(x3[c], Wc1[c * 16 + k], h);
    acc = fmaf(fmaxf(h, 0.f), Wc2[k], acc);
  }
  out[n] = acc;
}

// ---------------- host launcher ----------------
static inline int cdiv(long long a, int b) { return (int)((a + b - 1) / b); }

extern "C" void kernel_launch(void* const* d_in, const int* in_sizes, int n_in,
                              void* d_out, int out_size, void* d_ws, size_t ws_size,
                              hipStream_t stream) {
  const float* x    = (const float*)d_in[0];
  const int*   ei   = (const int*)  d_in[1];
  const float* W1   = (const float*)d_in[2];
  const float* aS1  = (const float*)d_in[3];
  const float* aD1  = (const float*)d_in[4];
  const float* b1   = (const float*)d_in[5];
  const float* W2   = (const float*)d_in[6];
  const float* aS2  = (const float*)d_in[7];
  const float* aD2  = (const float*)d_in[8];
  const float* b2   = (const float*)d_in[9];
  const float* Wc1  = (const float*)d_in[10];
  const float* bc1  = (const float*)d_in[11];
  const float* Wc2  = (const float*)d_in[12];
  const float* bc2  = (const float*)d_in[13];
  float* out = (float*)d_out;

  float* w = (float*)d_ws;
  float* h1   = w + OFF_H1;   float* as1 = w + OFF_AS1; float* ad1 = w + OFF_AD1;
  float* m1   = w + OFF_M1;   float* d1  = w + OFF_D1;  float* ex1 = w + OFF_EX1;
  float* out1 = w + OFF_OUT1;
  _Float16* x2h = (_Float16*)(w + OFF_X2H);
  _Float16* W2t = (_Float16*)(w + OFF_W2T);
  float* h2   = w + OFF_H2;   float* as2 = w + OFF_AS2; float* ad2 = w + OFF_AD2;
  float* m2   = w + OFF_M2;   float* d2  = w + OFF_D2;  float* ex2 = w + OFF_EX2;
  float* out2 = w + OFF_OUT2;

  const int B = 256;
  // ---- layer 1 ----
  k_lin1<<<cdiv((long long)NN * HEADS, B), B, 0, stream>>>(x, W1, aS1, aD1, h1, as1, ad1);
  k_init1<<<cdiv((long long)NN * F1, B), B, 0, stream>>>(out1, m1, d1);
  k_edge1_max<<<cdiv(ETOT, B), B, 0, stream>>>(ei, as1, ad1, m1);
  k_edge1_sum<<<cdiv(ETOT, B), B, 0, stream>>>(ei, as1, ad1, m1, ex1, d1);
  k_edge1_agg<<<cdiv((long long)ETOT * 32, B), B, 0, stream>>>(ei, ex1, d1, h1, out1);
  k_elu1<<<cdiv((long long)NN * F1, B), B, 0, stream>>>(out1, b1, x2h);
  // ---- layer 2 ----
  k_cvtW2<<<cdiv(F1 * HID, B), B, 0, stream>>>(W2, W2t);
  k_gemm2<<<cdiv((long long)(NN / 16) * (HID / 16) * 32, B), B, 0, stream>>>(x2h, W2t, h2);
  k_att2<<<cdiv(NN, B), B, 0, stream>>>(h2, aS2, aD2, as2, ad2);
  k_init2<<<cdiv((long long)NN * HID, B), B, 0, stream>>>(out2, m2, d2);
  k_edge2_max<<<cdiv(ETOT, B), B, 0, stream>>>(ei, as2, ad2, m2);
  k_edge2_sum<<<cdiv(ETOT, B), B, 0, stream>>>(ei, as2, ad2, m2, ex2, d2);
  k_edge2_agg<<<cdiv((long long)ETOT * 8, B), B, 0, stream>>>(ei, ex2, d2, h2, out2);
  // ---- head ----
  k_head<<<cdiv(NN, B), B, 0, stream>>>(out2, b2, Wc1, bc1, Wc2, bc2, out);
}
